// MultiHeadAttention_58737972740468
// MI455X (gfx1250) — compile-verified, matrix operand order checked by reading, not run on hardware
//
#include <hip/hip_runtime.h>
#include <hip/hip_bf16.h>

// ---------------------------------------------------------------------------
// Multi-head attention for MI455X (gfx1250), bf16 WMMA + f32 accumulate.
//   N=2, L=2048, E=1024, H=16, D=64.  scores scaled by 1/sqrt(E) = 1/32.
// ---------------------------------------------------------------------------

typedef __attribute__((ext_vector_type(16))) __bf16 v16bf;
typedef __attribute__((ext_vector_type(8)))  float  v8f;

#define EMBED 1024
#define HEADS 16
#define HDIM  64
#define SEQ   2048
#define BATCH 2
#define MROWS (BATCH * SEQ)     // 4096

#if __has_builtin(__builtin_amdgcn_global_load_async_to_lds_b128) && \
    __has_builtin(__builtin_amdgcn_s_wait_asynccnt)
#define HAVE_ASYNC_LDS 1
#else
#define HAVE_ASYNC_LDS 0
#endif

#define AS_GLOBAL __attribute__((address_space(1)))
#define AS_SHARED __attribute__((address_space(3)))

typedef int v4i_vs __attribute__((vector_size(16)));  // matches builtin param

// 16-byte global -> LDS copy; async (no VGPR roundtrip, ASYNCcnt) if available.
__device__ __forceinline__ void copy16_g2l(const __bf16* g, __bf16* l) {
#if HAVE_ASYNC_LDS
  __builtin_amdgcn_global_load_async_to_lds_b128(
      (AS_GLOBAL v4i_vs*)g, (AS_SHARED v4i_vs*)l, 0, 0);
#else
  *(uint4*)l = *(const uint4*)g;
#endif
}

__device__ __forceinline__ void drain_async() {
#if HAVE_ASYNC_LDS
  __builtin_amdgcn_s_wait_asynccnt(0);
#endif
}

union FragB {
  v16bf v;
  unsigned int u[8];
  __bf16 b[16];
};

// K-index inside a 16x32 bf16 A/B fragment for VGPR j (0..7), half = lane>>4.
__device__ __forceinline__ int kj_of(int j, int half) {
  return ((j >> 2) << 4) + (half << 3) + ((j & 3) << 1);
}

__device__ __forceinline__ v8f v8f_zero() {
  v8f z = {0.f, 0.f, 0.f, 0.f, 0.f, 0.f, 0.f, 0.f};
  return z;
}

// ---------------------------------------------------------------------------
// fp32 -> bf16 elementwise convert (vectorized x4)
// ---------------------------------------------------------------------------
__global__ void f32_to_bf16_kernel(const float* __restrict__ in,
                                   __bf16* __restrict__ out, int n4) {
  union Pack4 { __bf16 b[4]; ushort2 s2[2]; };
  int i = blockIdx.x * blockDim.x + threadIdx.x;
  int stride = gridDim.x * blockDim.x;
  for (; i < n4; i += stride) {
    float4 f = ((const float4*)in)[i];
    Pack4 p;
    p.b[0] = (__bf16)f.x; p.b[1] = (__bf16)f.y;
    p.b[2] = (__bf16)f.z; p.b[3] = (__bf16)f.w;
    ((ushort2*)out)[2 * i]     = p.s2[0];
    ((ushort2*)out)[2 * i + 1] = p.s2[1];
  }
}

// ---------------------------------------------------------------------------
// GEMM: Out[m][n] = sum_k A[m][k] * W[n][k]   (i.e. X @ W^T)
//   A: [MROWS][EMBED] bf16 row-major, W: [EMBED][EMBED] bf16 row-major (n,k)
//   MODE 0: bf16 output, head-split layout [(nbat*H + h)][SEQ][HDIM]
//   MODE 1: f32 output [MROWS][EMBED] with bias
// Block: 256 threads = 8 waves; block tile 128(M) x 64(N); K-step 64 via LDS
// (24 KB), 8 WMMAs per wave per barrier stage.
// ---------------------------------------------------------------------------
template <int MODE>
__global__ __launch_bounds__(256) void gemm_bf16_kernel(
    const __bf16* __restrict__ A, const __bf16* __restrict__ W,
    __bf16* __restrict__ outb, float* __restrict__ outf,
    const float* __restrict__ bias) {
  __shared__ __align__(16) __bf16 As[128 * 64];
  __shared__ __align__(16) __bf16 Bs[64 * 64];

  const int tid  = threadIdx.x;
  const int wave = tid >> 5;
  const int lane = tid & 31;
  const int half = lane >> 4;
  const int l15  = lane & 15;
  const int mb   = blockIdx.y * 128;
  const int nb   = blockIdx.x * 64;

  v8f acc[4];
  acc[0] = v8f_zero(); acc[1] = v8f_zero();
  acc[2] = v8f_zero(); acc[3] = v8f_zero();

  const int ar = tid >> 1, ac = (tid & 1) * 32;  // A: 32 elems / thread
  const int br = tid >> 2, bc = (tid & 3) * 16;  // B: 16 elems / thread

  for (int kk = 0; kk < EMBED; kk += 64) {
    const __bf16* gA = A + (size_t)(mb + ar) * EMBED + kk + ac;
    const __bf16* gB = W + (size_t)(nb + br) * EMBED + kk + bc;
#pragma unroll
    for (int c = 0; c < 4; ++c)
      copy16_g2l(gA + 8 * c, &As[ar * 64 + ac + 8 * c]);
#pragma unroll
    for (int c = 0; c < 2; ++c)
      copy16_g2l(gB + 8 * c, &Bs[br * 64 + bc + 8 * c]);
    if (kk + 64 < EMBED) {
      __builtin_prefetch(gA + 64, 0, 1);
      __builtin_prefetch(gB + 64, 0, 1);
    }
    drain_async();
    __syncthreads();

#pragma unroll
    for (int koff = 0; koff < 64; koff += 32) {
      FragB af;
#pragma unroll
      for (int j = 0; j < 8; ++j)
        af.u[j] = *(const unsigned int*)
            &As[(wave * 16 + l15) * 64 + koff + kj_of(j, half)];
#pragma unroll
      for (int s = 0; s < 4; ++s) {
        FragB bfr;
#pragma unroll
        for (int j = 0; j < 8; ++j)
          bfr.u[j] = *(const unsigned int*)
              &Bs[(s * 16 + l15) * 64 + koff + kj_of(j, half)];
        acc[s] = __builtin_amdgcn_wmma_f32_16x16x32_bf16(
            false, af.v, false, bfr.v, (short)0, acc[s], false, false);
      }
    }
    __syncthreads();
  }

#pragma unroll
  for (int s = 0; s < 4; ++s) {
    const int ncol = nb + s * 16 + l15;
#pragma unroll
    for (int j = 0; j < 8; ++j) {
      const int m = mb + wave * 16 + half * 8 + j;
      if (MODE == 0) {
        const int nbat = m >> 11, l = m & (SEQ - 1);
        const int h = ncol >> 6, d = ncol & 63;
        outb[((size_t)(nbat * HEADS + h) * SEQ + l) * HDIM + d] =
            (__bf16)acc[s][j];
      } else {
        outf[(size_t)m * EMBED + ncol] = acc[s][j] + bias[ncol];
      }
    }
  }
}

// ---------------------------------------------------------------------------
// Flash attention. Q/K/V head-major: base = (nbat*H + h)*SEQ*HDIM.
// Block: 256 threads = 8 waves; each wave owns 16 query rows; all 8 waves
// share each 32-key K/V tile via LDS.  Per 32-key tile:
//   S^T: two 16k x 16q tiles = K_tile(A, K-dim=d) x Q^T(B)      [4 WMMAs]
//   online softmax per query (query index = lane, shfl_xor(16) combine)
//   O^T(64d x 16q) += V^T(A) x P^T(B) with full K=32            [4 WMMAs]
// P^T C-layout maps element-for-element onto the B operand: elements 0..7
// come from score tile 0 (keys half*8+0..7), 8..15 from tile 1 (keys
// 16+half*8+0..7).  K tile streams in via async-to-LDS; V is staged
// TRANSPOSED (Vt[d][key]) so each A-fragment dword is one ds_load_b32.
// ---------------------------------------------------------------------------
__global__ __launch_bounds__(256) void attention_kernel(
    const __bf16* __restrict__ Qh, const __bf16* __restrict__ Kh,
    const __bf16* __restrict__ Vh, __bf16* __restrict__ Ob) {
  __shared__ __align__(16) __bf16 Ks[32 * 64];   // [key][d]
  __shared__ __align__(16) __bf16 Vt[64 * 32];   // [d][key]  (transposed)

  const int tid  = threadIdx.x;
  const int wave = tid >> 5;
  const int lane = tid & 31;
  const int half = lane >> 4;
  const int l15  = lane & 15;
  const int hd   = blockIdx.y;                    // nbat*HEADS + h
  const int q0   = blockIdx.x * 128 + wave * 16;  // first query row of wave
  const size_t base = (size_t)hd * SEQ * HDIM;

  // Q^T fragments (B operand), K-dim = d. Pairs (d, d+1) are contiguous.
  FragB bq0, bq1;
#pragma unroll
  for (int j = 0; j < 8; ++j) {
    const __bf16* qrow = Qh + base + (size_t)(q0 + l15) * HDIM;
    bq0.u[j] = *(const unsigned int*)(qrow + kj_of(j, half));
    bq1.u[j] = *(const unsigned int*)(qrow + 32 + kj_of(j, half));
  }

  float mrun = -1e30f, lrun = 0.0f;
  v8f o[4];
  o[0] = v8f_zero(); o[1] = v8f_zero(); o[2] = v8f_zero(); o[3] = v8f_zero();

  const int tr = tid >> 3;        // 0..31: key row of this thread's uint4
  const int tc = (tid & 7) * 8;   // 0..56: d offset (8 bf16 per uint4)

  for (int kt = 0; kt < SEQ / 32; ++kt) {
    // K tile row-major via async DMA; V tile transposed into Vt[d][key].
    copy16_g2l(Kh + base + (size_t)(kt * 32 + tr) * HDIM + tc,
               &Ks[tr * 64 + tc]);
    union { uint4 q; __bf16 b[8]; } vld;
    vld.q = *(const uint4*)(Vh + base + (size_t)(kt * 32 + tr) * HDIM + tc);
#pragma unroll
    for (int e = 0; e < 8; ++e) Vt[(tc + e) * 32 + tr] = vld.b[e];
    if (kt + 1 < SEQ / 32) {
      __builtin_prefetch(Kh + base + (size_t)(kt * 32 + 32 + tr) * HDIM + tc, 0, 1);
      __builtin_prefetch(Vh + base + (size_t)(kt * 32 + 32 + tr) * HDIM + tc, 0, 1);
    }
    drain_async();
    __syncthreads();

    // Score tiles: A = K rows (M = key, K-dim = d), B = Q^T.
    FragB aka, akb;
    v8f s0, s1;
#pragma unroll
    for (int j = 0; j < 8; ++j) {
      aka.u[j] = *(const unsigned int*)&Ks[l15 * 64 + kj_of(j, half)];
      akb.u[j] = *(const unsigned int*)&Ks[l15 * 64 + 32 + kj_of(j, half)];
    }
    s0 = __builtin_amdgcn_wmma_f32_16x16x32_bf16(
        false, aka.v, false, bq0.v, (short)0, v8f_zero(), false, false);
    s0 = __builtin_amdgcn_wmma_f32_16x16x32_bf16(
        false, akb.v, false, bq1.v, (short)0, s0, false, false);
#pragma unroll
    for (int j = 0; j < 8; ++j) {
      aka.u[j] = *(const unsigned int*)&Ks[(16 + l15) * 64 + kj_of(j, half)];
      akb.u[j] = *(const unsigned int*)&Ks[(16 + l15) * 64 + 32 + kj_of(j, half)];
    }
    s1 = __builtin_amdgcn_wmma_f32_16x16x32_bf16(
        false, aka.v, false, bq0.v, (short)0, v8f_zero(), false, false);
    s1 = __builtin_amdgcn_wmma_f32_16x16x32_bf16(
        false, akb.v, false, bq1.v, (short)0, s1, false, false);

    // Online softmax over 32 keys (16 per lane + shfl_xor(16) combine).
    float p[16];
    float mx = -1e30f;
#pragma unroll
    for (int j = 0; j < 8; ++j) {
      p[j]     = s0[j] * 0.03125f;   // 1/sqrt(EMBED)
      p[8 + j] = s1[j] * 0.03125f;
      mx = fmaxf(mx, fmaxf(p[j], p[8 + j]));
    }
    mx = fmaxf(mx, __shfl_xor(mx, 16));
    const float mnew  = fmaxf(mrun, mx);
    const float alpha = __expf(mrun - mnew);
    float ps = 0.0f;
#pragma unroll
    for (int j = 0; j < 16; ++j) {
      p[j] = __expf(p[j] - mnew);
      ps += p[j];
    }
    ps += __shfl_xor(ps, 16);
    lrun = lrun * alpha + ps;
    mrun = mnew;
#pragma unroll
    for (int s4 = 0; s4 < 4; ++s4) o[s4] = o[s4] * alpha;

    // P^T as B operand: element i <-> key (i<8 ? half*8+i : 16+half*8+i-8).
    FragB bp;
#pragma unroll
    for (int j = 0; j < 16; ++j) bp.b[j] = (__bf16)p[j];

    // V^T fragments (A operand) from transposed LDS: one dword per VGPR.
#pragma unroll
    for (int s4 = 0; s4 < 4; ++s4) {
      FragB av;
#pragma unroll
      for (int j = 0; j < 8; ++j)
        av.u[j] =
            *(const unsigned int*)&Vt[(s4 * 16 + l15) * 32 + kj_of(j, half)];
      o[s4] = __builtin_amdgcn_wmma_f32_16x16x32_bf16(
          false, av.v, false, bp.v, (short)0, o[s4], false, false);
    }
    __syncthreads();
  }

  // Epilogue: O^T C-layout: lane -> q, rows -> d. Write [MROWS][EMBED] bf16.
  const float invl = 1.0f / lrun;
  const int nbat = hd >> 4, h = hd & (HEADS - 1);
  const int q = q0 + l15;
  const size_t rowbase = (size_t)(nbat * SEQ + q) * EMBED + h * HDIM;
#pragma unroll
  for (int s4 = 0; s4 < 4; ++s4)
#pragma unroll
    for (int j = 0; j < 8; ++j)
      Ob[rowbase + s4 * 16 + half * 8 + j] = (__bf16)(o[s4][j] * invl);
}

// ---------------------------------------------------------------------------
// Host launcher
// ---------------------------------------------------------------------------
extern "C" void kernel_launch(void* const* d_in, const int* in_sizes, int n_in,
                              void* d_out, int out_size, void* d_ws,
                              size_t ws_size, hipStream_t stream) {
  (void)in_sizes; (void)n_in; (void)out_size; (void)ws_size;

  const float* x  = (const float*)d_in[0];
  const float* Wq = (const float*)d_in[1];
  const float* Wk = (const float*)d_in[2];
  const float* Wv = (const float*)d_in[3];
  const float* Wo = (const float*)d_in[4];
  const float* bo = (const float*)d_in[5];
  float* out = (float*)d_out;

  char* ws = (char*)d_ws;
  size_t off = 0;
  auto alloc = [&](size_t bytes) {
    char* p = ws + off;
    off += (bytes + 255) & ~(size_t)255;
    return p;
  };
  const size_t xbytes = (size_t)MROWS * EMBED * sizeof(__bf16);
  const size_t wbytes = (size_t)EMBED * EMBED * sizeof(__bf16);
  __bf16* xb  = (__bf16*)alloc(xbytes);
  __bf16* wqb = (__bf16*)alloc(wbytes);
  __bf16* wkb = (__bf16*)alloc(wbytes);
  __bf16* wvb = (__bf16*)alloc(wbytes);
  __bf16* wob = (__bf16*)alloc(wbytes);
  __bf16* Qh  = (__bf16*)alloc(xbytes);
  __bf16* Kh  = (__bf16*)alloc(xbytes);
  __bf16* Vh  = (__bf16*)alloc(xbytes);
  __bf16* Ab  = (__bf16*)alloc(xbytes);

  f32_to_bf16_kernel<<<1024, 256, 0, stream>>>(x, xb, MROWS * EMBED / 4);
  f32_to_bf16_kernel<<<1024, 256, 0, stream>>>(Wq, wqb, EMBED * EMBED / 4);
  f32_to_bf16_kernel<<<1024, 256, 0, stream>>>(Wk, wkb, EMBED * EMBED / 4);
  f32_to_bf16_kernel<<<1024, 256, 0, stream>>>(Wv, wvb, EMBED * EMBED / 4);
  f32_to_bf16_kernel<<<1024, 256, 0, stream>>>(Wo, wob, EMBED * EMBED / 4);

  dim3 ggrid(EMBED / 64, MROWS / 128);
  gemm_bf16_kernel<0><<<ggrid, 256, 0, stream>>>(xb, wqb, Qh, nullptr, nullptr);
  gemm_bf16_kernel<0><<<ggrid, 256, 0, stream>>>(xb, wkb, Kh, nullptr, nullptr);
  gemm_bf16_kernel<0><<<ggrid, 256, 0, stream>>>(xb, wvb, Vh, nullptr, nullptr);

  attention_kernel<<<dim3(SEQ / 128, BATCH * HEADS), 256, 0, stream>>>(
      Qh, Kh, Vh, Ab);

  gemm_bf16_kernel<1><<<ggrid, 256, 0, stream>>>(Ab, wob, nullptr, out, bo);
}
